// RNNAttention_New_9706626090130
// MI455X (gfx1250) — matrix operand order repaired and checked by reading, not compiled, not run on hardware
//
#include <hip/hip_runtime.h>
#include <math.h>
#include <stdint.h>

// ---------------------------------------------------------------------------
// RNN(GRU)+attention LM for MI455X (gfx1250, wave32, WMMA bf16->f32).
// - All GEMMs: v_wmma_f32_16x16x32_bf16, fp32 accumulate.
// - GEMM operand staging: Tensor Data Mover (tensor_load_to_lds) into LDS,
//   double buffered, tracked with TENSORcnt.
// - GRU scan: persistent 64-WG kernel, W_hh slice resident in LDS (bf16),
//   per-step WMMA + global atomic barrier.
// Workspace use ~279 MB.
// ---------------------------------------------------------------------------

typedef unsigned short u16;
typedef __attribute__((ext_vector_type(16))) __bf16 v16bf;
typedef __attribute__((ext_vector_type(8)))  float  v8f;
typedef u16 us8 __attribute__((ext_vector_type(8)));
typedef unsigned int v4u __attribute__((ext_vector_type(4)));
typedef int v8i __attribute__((ext_vector_type(8)));
typedef int v4i __attribute__((ext_vector_type(4)));

#define Bb 4
#define Tt 1024
#define Ee 512
#define Hh 1024
#define Vv 32000
#define MROWS (Bb * Tt)               // 4096
#define OUT_LOGITS ((long)MROWS * Vv) // then hT [B,H]

__device__ __forceinline__ u16 f2bf(float f) {
  unsigned u = __float_as_uint(f);
  u += 0x7fffu + ((u >> 16) & 1u); // round-to-nearest-even
  return (u16)(u >> 16);
}
__device__ __forceinline__ float bf2f(u16 h) {
  return __uint_as_float(((unsigned)h) << 16);
}

// Build a 16-half fragment from two contiguous 16-byte chunks.
__device__ __forceinline__ v16bf load_frag(const u16* p0, const u16* p1) {
  union { us8 h[2]; v16bf v; } x;
  x.h[0] = *(const us8*)p0;
  x.h[1] = *(const us8*)p1;
  return x.v;
}
__device__ __forceinline__ v8f wmma_bf16(v16bf a, v16bf b, v8f c) {
  return __builtin_amdgcn_wmma_f32_16x16x32_bf16(false, a, false, b,
                                                 (short)0, c, false, false);
}

// LDS byte offset of a __shared__ object: flat LDS-aperture addresses carry
// the LDS offset in the low 32 bits (ISA 10.2).
__device__ __forceinline__ unsigned lds_off32(const void* p) {
  return (unsigned)(uintptr_t)p;
}

// ---------------------------------------------------------------------------
// TDM: DMA a 2-D bf16 tile [rows x 64] (row stride = row_stride_elems) from
// global memory into LDS at lds_off, packed row-major with pitch 64.
// D# layout per CDNA5 ISA 8.3/8.4: group0 {count=1, lds_addr, global_addr,
// type=2}; group1 {data_size=2B, tensor dims (large), tile_dim0=64,
// tile_dim1=rows, tensor_dim0_stride}. Groups 2/3 zero (2-D tile).
// Issued by one wave; completion tracked with TENSORcnt.
// ---------------------------------------------------------------------------
#define HAVE_TDM __has_builtin(__builtin_amdgcn_tensor_load_to_lds)

__device__ __forceinline__ void tdm_load_tile(unsigned lds_off,
                                              const u16* gptr, int rows,
                                              int row_stride_elems) {
#if HAVE_TDM
  unsigned long long ga = (unsigned long long)(uintptr_t)gptr;
  v4u g0;
  g0[0] = 1u;                                 // count=1, user mode
  g0[1] = lds_off;                            // lds_addr (bytes)
  g0[2] = (unsigned)ga;                       // global_addr[31:0]
  g0[3] = (unsigned)(ga >> 32) | (2u << 30);  // global_addr[56:32] | type=2
  const unsigned long long td0 = 0x7FFFFFFFull;        // tensor_dim0 (no OOB)
  const unsigned long long td1 = 0x7FFFFFFFull;        // tensor_dim1 (no OOB)
  const unsigned long long st = (unsigned long long)row_stride_elems;
  v8i g1;
  g1[0] = (int)(1u << 16);                               // data_size=1 (2 B)
  g1[1] = (int)((td0 & 0xFFFFull) << 16);                // td0[15:0]
  g1[2] = (int)(((td0 >> 16) & 0xFFFFull) | ((td1 & 0xFFFFull) << 16));
  g1[3] = (int)(((td1 >> 16) & 0xFFFFull) | (64u << 16));  // tile_dim0=64
  g1[4] = (int)(rows & 0xFFFF);                            // tile_dim1, d2=0
  g1[5] = (int)(st & 0xFFFFFFFFull);                       // stride[31:0]
  g1[6] = (int)((st >> 32) & 0xFFFFull);                   // stride[47:32]
  g1[7] = 0;
  v4i g2 = {0, 0, 0, 0};
  v4i g3 = {0, 0, 0, 0};
#if __clang_major__ >= 23
  v8i g4 = {0, 0, 0, 0, 0, 0, 0, 0};
  __builtin_amdgcn_tensor_load_to_lds(g0, g1, g2, g3, g4, 0);
#else
  __builtin_amdgcn_tensor_load_to_lds(g0, g1, g2, g3, 0);
#endif
#else
  // Fallback: no copy is done; the tile is left unloaded.
  u16* dst = (u16*)((char*)nullptr); // unreachable when HAVE_TDM
  (void)dst; (void)lds_off; (void)gptr; (void)rows; (void)row_stride_elems;
#endif
}

// ---------------------------------------------------------------------------
// Generic bf16 WMMA GEMM:  C[M,N] = scale * (A[M,K] @ W[N,K]^T) (+bias[n])
// grid = (M/64, N/256, batch); block = 256 (8 waves: 2 M-slabs x 4 N-slabs).
// K is consumed in 64-wide chunks; A (64x64) and W (256x64) chunk tiles are
// TDM-DMA'd into double-buffered LDS; fragments then ds_load_b128 with
// k = 16*(lane/16) + s in A and B. One workgroup barrier per chunk; DMA of chunk i+1 overlaps
// WMMA on chunk i.
// ---------------------------------------------------------------------------
template <int OUT_BF16, int HAS_BIAS, int CAUSAL>
__global__ __launch_bounds__(256) void gemm_bf16_wmma(
    const u16* __restrict__ A, const u16* __restrict__ W,
    const float* __restrict__ bias, void* __restrict__ Cout,
    int K, int ldc, int col_off,
    long strideA, long strideW, long strideC, float scale) {
  __shared__ __align__(16) u16 Wt[2][256 * 64]; // 64 KB
  __shared__ __align__(16) u16 At[2][64 * 64];  // 16 KB
  const int z = blockIdx.z;
  A += (long)z * strideA;
  W += (long)z * strideW;
  const int m0   = blockIdx.x * 64;
  const int n0   = blockIdx.y * 256;
  const int wave = threadIdx.x >> 5;
  const int lane = threadIdx.x & 31;
  const int wm = wave & 1, wn = wave >> 1;
  const int lrow = lane & 15, hi = lane >> 4;

  const u16* Ablk = A + (size_t)m0 * K;
  const u16* Wblk = W + (size_t)n0 * K;
  const int nchunks = K >> 6;

  if (wave == 0) {
    tdm_load_tile(lds_off32(&At[0][0]), Ablk, 64, K);
    tdm_load_tile(lds_off32(&Wt[0][0]), Wblk, 256, K);
  }

  const v8f cz = {0.f, 0.f, 0.f, 0.f, 0.f, 0.f, 0.f, 0.f};
  v8f c[2][4];
#pragma unroll
  for (int i = 0; i < 2; ++i)
#pragma unroll
    for (int f = 0; f < 4; ++f) c[i][f] = cz;

  for (int ch = 0; ch < nchunks; ++ch) {
    const int cur = ch & 1;
    if (wave == 0) __builtin_amdgcn_s_wait_tensorcnt(0);
    __syncthreads(); // chunk `ch` resident; previous readers of buf cur^1 done
    if (wave == 0 && ch + 1 < nchunks) {
      const int k0n = (ch + 1) << 6;
      tdm_load_tile(lds_off32(&At[cur ^ 1][0]), Ablk + k0n, 64, K);
      tdm_load_tile(lds_off32(&Wt[cur ^ 1][0]), Wblk + k0n, 256, K);
    }
    const u16* As = &At[cur][0];
    const u16* Ws = &Wt[cur][0];
#pragma unroll
    for (int ks = 0; ks < 2; ++ks) {
      const u16* ap0 = As + (wm * 32 + lrow) * 64 + ks * 32 + hi * 16;
      v16bf a0 = load_frag(ap0, ap0 + 8);
      const u16* ap1 = ap0 + 16 * 64;
      v16bf a1 = load_frag(ap1, ap1 + 8);
#pragma unroll
      for (int f = 0; f < 4; ++f) {
        const u16* bp = Ws + (wn * 64 + f * 16 + lrow) * 64 + ks * 32 + hi * 16;
        v16bf b = load_frag(bp, bp + 8);
        c[0][f] = wmma_bf16(a0, b, c[0][f]);
        c[1][f] = wmma_bf16(a1, b, c[1][f]);
      }
    }
  }

#pragma unroll
  for (int fm = 0; fm < 2; ++fm)
#pragma unroll
    for (int f = 0; f < 4; ++f)
#pragma unroll
      for (int r = 0; r < 8; ++r) {
        int m = m0 + wm * 32 + fm * 16 + r + hi * 8;
        int n = n0 + wn * 64 + f * 16 + lrow;
        float v = c[fm][f][r] * scale;
        if (HAS_BIAS) v += bias[n];
        if (CAUSAL) { if (n > m) v = -3.0e38f; }
        size_t ci = (size_t)z * strideC + (size_t)m * ldc + col_off + n;
        if (OUT_BF16) ((u16*)Cout)[ci] = f2bf(v);
        else          ((float*)Cout)[ci] = v;
      }
}

// ---------------------------------------------------------------------------
// GRU scan: 64 persistent WGs. WG g owns hidden units j in [g*16, g*16+16).
// LDS: bf16 W slice (48 rows: r,z,n gates x 16 units), bf16 h (rows 4..15
// zero-padded for the 16-row WMMA A fragment), per-wave C partials.
// Per step: 8 waves split K=1024 (128 each), 3 N-fragments -> 12 WMMAs,
// LDS reduction, gate math on 64 threads, global h publish, atomic barrier.
// ---------------------------------------------------------------------------
#define GWG 64
__global__ __launch_bounds__(256) void gru_scan(
    const float* __restrict__ Whh, const float* __restrict__ bhh,
    const float* __restrict__ xp, float* __restrict__ hglob,
    u16* __restrict__ gru_bf, float* __restrict__ hT_out,
    unsigned* __restrict__ cnt) {
  __shared__ __align__(16) u16 Wl[48 * 1024]; // 98304 B
  __shared__ __align__(16) u16 hl[16 * 1024]; // 32768 B
  __shared__ float part[8][3][256];           // 24576 B
  __shared__ float hself[64];                 //   256 B
  const int g = blockIdx.x, tid = threadIdx.x;
  const int wave = tid >> 5, lane = tid & 31;
  const int lrow = lane & 15, hi = lane >> 4;

  for (int i = tid; i < 48 * 1024; i += 256) {
    int lr = i >> 10, k = i & 1023;
    int gate = lr >> 4, jl = lr & 15;
    Wl[i] = f2bf(Whh[(size_t)(gate * Hh + g * 16 + jl) * Hh + k]);
  }
  for (int i = tid; i < 16 * 1024; i += 256) hl[i] = 0;
  if (tid < 64) hself[tid] = 0.f;
  __syncthreads();

  const v8f cz = {0.f, 0.f, 0.f, 0.f, 0.f, 0.f, 0.f, 0.f};
  const int kw = wave * 128;

  for (int t = 0; t < Tt; ++t) {
    v8f c0 = cz, c1 = cz, c2 = cz;
#pragma unroll
    for (int ks = 0; ks < 4; ++ks) {
      int k0 = kw + ks * 32;
      const u16* ha = &hl[lrow * 1024 + k0 + hi * 16];
      v16bf a = load_frag(ha, ha + 8);
      const u16* b0p = &Wl[(0 * 16 + lrow) * 1024 + k0 + hi * 16];
      const u16* b1p = &Wl[(1 * 16 + lrow) * 1024 + k0 + hi * 16];
      const u16* b2p = &Wl[(2 * 16 + lrow) * 1024 + k0 + hi * 16];
      c0 = wmma_bf16(a, load_frag(b0p, b0p + 8), c0);
      c1 = wmma_bf16(a, load_frag(b1p, b1p + 8), c1);
      c2 = wmma_bf16(a, load_frag(b2p, b2p + 8), c2);
    }
#pragma unroll
    for (int r = 0; r < 8; ++r) {
      part[wave][0][r * 32 + lane] = c0[r];
      part[wave][1][r * 32 + lane] = c1[r];
      part[wave][2][r * 32 + lane] = c2[r];
    }
    __syncthreads();

    if (tid < 64) {
      int b = tid >> 4, jl = tid & 15;
      float s[3];
#pragma unroll
      for (int gate = 0; gate < 3; ++gate) {
        float acc = bhh[gate * Hh + g * 16 + jl];
#pragma unroll
        for (int w = 0; w < 8; ++w) acc += part[w][gate][b * 32 + jl];
        s[gate] = acc;
      }
      const float* xprow = xp + ((size_t)b * Tt + t) * (3 * Hh) + g * 16 + jl;
      float xr = xprow[0], xz = xprow[Hh], xn = xprow[2 * Hh];
      float r_ = 1.f / (1.f + expf(-(xr + s[0])));
      float z_ = 1.f / (1.f + expf(-(xz + s[1])));
      float n_ = tanhf(xn + r_ * s[2]);
      float h2 = (1.f - z_) * n_ + z_ * hself[tid];
      hself[tid] = h2;
      hglob[b * Hh + g * 16 + jl] = h2;
      gru_bf[((size_t)b * Tt + t) * Hh + g * 16 + jl] = f2bf(h2);
    }
    __threadfence();
    __syncthreads();
    if (tid == 0) {
      atomicAdd(cnt, 1u);
      unsigned target = (unsigned)(t + 1) * GWG;
      while (__hip_atomic_load(cnt, __ATOMIC_RELAXED,
                               __HIP_MEMORY_SCOPE_AGENT) < target)
        __builtin_amdgcn_s_sleep(8);
    }
    __syncthreads();
    __threadfence();
    for (int i = tid; i < Bb * Hh; i += 256) hl[i] = f2bf(hglob[i]);
    __syncthreads();
  }
  if (tid < 64)
    hT_out[(tid >> 4) * Hh + g * 16 + (tid & 15)] = hself[tid];
}

// ---------------------------------------------------------------------------
// Small helper kernels
// ---------------------------------------------------------------------------
__global__ void init_state(float* hglob, unsigned* cnt) {
  int i = blockIdx.x * blockDim.x + threadIdx.x;
  if (i < Bb * Hh) hglob[i] = 0.f;
  if (i == 0) *cnt = 0u;
}
__global__ void cvt_bf16(const float* __restrict__ src, u16* __restrict__ dst,
                         long n) {
  long i = (long)blockIdx.x * blockDim.x + threadIdx.x;
  long stride = (long)gridDim.x * blockDim.x;
  for (; i < n; i += stride) dst[i] = f2bf(src[i]);
}
__global__ void embed_gather(const int* __restrict__ x,
                             const float* __restrict__ emb,
                             u16* __restrict__ e_bf) {
  int row = blockIdx.x;
  const float* src = emb + (size_t)x[row] * Ee;
  u16* dst = e_bf + (size_t)row * Ee;
  for (int k = threadIdx.x; k < Ee; k += 256) dst[k] = f2bf(src[k]);
}
__global__ void transpose_V(const u16* __restrict__ in, u16* __restrict__ out) {
  __shared__ u16 tile[32][33];
  int b = blockIdx.z, t0 = blockIdx.x * 32, h0 = blockIdx.y * 32;
  int tx = threadIdx.x & 31, ty8 = threadIdx.x >> 5;
  const u16* src = in + ((size_t)b << 20);
  u16* dst = out + ((size_t)b << 20);
#pragma unroll
  for (int i = 0; i < 4; ++i) {
    int ty = ty8 + i * 8;
    tile[ty][tx] = src[(size_t)(t0 + ty) * Hh + h0 + tx];
  }
  __syncthreads();
#pragma unroll
  for (int i = 0; i < 4; ++i) {
    int ty = ty8 + i * 8;
    dst[(size_t)(h0 + ty) * Tt + t0 + tx] = tile[tx][ty];
  }
}
__global__ void concat_gru(const u16* __restrict__ gru_bf,
                           u16* __restrict__ combined) {
  long i = (long)blockIdx.x * blockDim.x + threadIdx.x;
  long stride = (long)gridDim.x * blockDim.x;
  for (; i < (long)MROWS * Hh; i += stride)
    combined[((i >> 10) << 11) + (i & 1023)] = gru_bf[i];
}
__global__ __launch_bounds__(256) void attn_softmax(const float* __restrict__ S,
                                                    u16* __restrict__ P) {
  __shared__ float red[256];
  int row = blockIdx.x, tid = threadIdx.x;
  const float* s = S + (size_t)row * Tt;
  float mx = -3.4e38f;
  for (int k = tid; k < Tt; k += 256) mx = fmaxf(mx, s[k]);
  red[tid] = mx; __syncthreads();
  for (int o = 128; o > 0; o >>= 1) {
    if (tid < o) red[tid] = fmaxf(red[tid], red[tid + o]);
    __syncthreads();
  }
  mx = red[0]; __syncthreads();
  float sum = 0.f;
  for (int k = tid; k < Tt; k += 256) sum += expf(s[k] - mx);
  red[tid] = sum; __syncthreads();
  for (int o = 128; o > 0; o >>= 1) {
    if (tid < o) red[tid] += red[tid + o];
    __syncthreads();
  }
  float inv = 1.f / red[0];
  for (int k = tid; k < Tt; k += 256)
    P[(size_t)row * Tt + k] = f2bf(expf(s[k] - mx) * inv);
}
__global__ __launch_bounds__(256) void logsoftmax_inplace(float* __restrict__ o) {
  __shared__ float red[256];
  int row = blockIdx.x, tid = threadIdx.x;
  float* p = o + (size_t)row * Vv;
  float mx = -3.4e38f;
  for (int k = tid; k < Vv; k += 256) mx = fmaxf(mx, p[k]);
  red[tid] = mx; __syncthreads();
  for (int q = 128; q > 0; q >>= 1) {
    if (tid < q) red[tid] = fmaxf(red[tid], red[tid + q]);
    __syncthreads();
  }
  mx = red[0]; __syncthreads();
  float sum = 0.f;
  for (int k = tid; k < Vv; k += 256) sum += expf(p[k] - mx);
  red[tid] = sum; __syncthreads();
  for (int q = 128; q > 0; q >>= 1) {
    if (tid < q) red[tid] += red[tid + q];
    __syncthreads();
  }
  float lse = mx + logf(red[0]);
  for (int k = tid; k < Vv; k += 256) p[k] = p[k] - lse;
}

// ---------------------------------------------------------------------------
static inline char* wsal(char*& p, size_t bytes) {
  char* r = p;
  p += (bytes + 255) & ~(size_t)255;
  return r;
}

extern "C" void kernel_launch(void* const* d_in, const int* in_sizes, int n_in,
                              void* d_out, int out_size, void* d_ws,
                              size_t ws_size, hipStream_t stream) {
  (void)in_sizes; (void)n_in; (void)out_size; (void)ws_size;
  const int*   x    = (const int*)d_in[0];
  const float* emb  = (const float*)d_in[1];
  const float* W_ih = (const float*)d_in[2];
  const float* b_ih = (const float*)d_in[3];
  const float* W_hh = (const float*)d_in[4];
  const float* b_hh = (const float*)d_in[5];
  const float* Wq   = (const float*)d_in[6];
  const float* bq   = (const float*)d_in[7];
  const float* Wk   = (const float*)d_in[8];
  const float* bk   = (const float*)d_in[9];
  const float* Wv   = (const float*)d_in[10];
  const float* bv   = (const float*)d_in[11];
  const float* W_fc = (const float*)d_in[12];
  const float* b_fc = (const float*)d_in[13];
  float* out = (float*)d_out;

  char* p = (char*)d_ws;
  u16*   e_bf    = (u16*)  wsal(p, (size_t)MROWS * Ee * 2);
  u16*   Wih_bf  = (u16*)  wsal(p, (size_t)3 * Hh * Ee * 2);
  float* xp      = (float*)wsal(p, (size_t)MROWS * 3 * Hh * 4);
  u16*   gru_bf  = (u16*)  wsal(p, (size_t)MROWS * Hh * 2);
  u16*   Wq_bf   = (u16*)  wsal(p, (size_t)Hh * Hh * 2);
  u16*   Wk_bf   = (u16*)  wsal(p, (size_t)Hh * Hh * 2);
  u16*   Wv_bf   = (u16*)  wsal(p, (size_t)Hh * Hh * 2);
  u16*   Qb      = (u16*)  wsal(p, (size_t)MROWS * Hh * 2);
  u16*   Kb      = (u16*)  wsal(p, (size_t)MROWS * Hh * 2);
  u16*   Vb      = (u16*)  wsal(p, (size_t)MROWS * Hh * 2);
  u16*   Vt      = (u16*)  wsal(p, (size_t)MROWS * Hh * 2);
  float* S       = (float*)wsal(p, (size_t)Bb * Tt * Tt * 4);
  u16*   P       = (u16*)  wsal(p, (size_t)Bb * Tt * Tt * 2);
  u16*   combined= (u16*)  wsal(p, (size_t)MROWS * 2 * Hh * 2);
  u16*   Wfc_bf  = (u16*)  wsal(p, (size_t)Vv * 2 * Hh * 2);
  float* hglob   = (float*)wsal(p, (size_t)Bb * Hh * 4);
  unsigned* cnt  = (unsigned*)wsal(p, 256);

  // 0) state + weight conversion to bf16
  init_state<<<16, 256, 0, stream>>>(hglob, cnt);
  cvt_bf16<<<1024, 256, 0, stream>>>(W_ih, Wih_bf, (long)3 * Hh * Ee);
  cvt_bf16<<<512, 256, 0, stream>>>(Wq, Wq_bf, (long)Hh * Hh);
  cvt_bf16<<<512, 256, 0, stream>>>(Wk, Wk_bf, (long)Hh * Hh);
  cvt_bf16<<<512, 256, 0, stream>>>(Wv, Wv_bf, (long)Hh * Hh);
  cvt_bf16<<<4096, 256, 0, stream>>>(W_fc, Wfc_bf, (long)Vv * 2 * Hh);

  // 1) embedding gather (bf16)
  embed_gather<<<MROWS, 256, 0, stream>>>(x, emb, e_bf);

  // 2) xp = e @ W_ih^T + b_ih     [4096,3072], K=512
  gemm_bf16_wmma<0, 1, 0><<<dim3(MROWS / 64, (3 * Hh) / 256, 1), 256, 0,
                            stream>>>(e_bf, Wih_bf, b_ih, xp, Ee, 3 * Hh, 0,
                                      0, 0, 0, 1.f);

  // 3) GRU scan (persistent, 64 WGs); writes gru_bf + hT tail of d_out
  gru_scan<<<GWG, 256, 0, stream>>>(W_hh, b_hh, xp, hglob, gru_bf,
                                    out + OUT_LOGITS, cnt);

  // 4) Q/K/V projections (bf16 out, bias)
  gemm_bf16_wmma<1, 1, 0><<<dim3(MROWS / 64, Hh / 256, 1), 256, 0, stream>>>(
      gru_bf, Wq_bf, bq, Qb, Hh, Hh, 0, 0, 0, 0, 1.f);
  gemm_bf16_wmma<1, 1, 0><<<dim3(MROWS / 64, Hh / 256, 1), 256, 0, stream>>>(
      gru_bf, Wk_bf, bk, Kb, Hh, Hh, 0, 0, 0, 0, 1.f);
  gemm_bf16_wmma<1, 1, 0><<<dim3(MROWS / 64, Hh / 256, 1), 256, 0, stream>>>(
      gru_bf, Wv_bf, bv, Vb, Hh, Hh, 0, 0, 0, 0, 1.f);

  // 5) V^T for the PV GEMM
  transpose_V<<<dim3(32, 32, Bb), 256, 0, stream>>>(Vb, Vt);

  // 6) scores = Q K^T * (1/32), causal mask   [per-batch 1024x1024]
  gemm_bf16_wmma<0, 0, 1><<<dim3(Tt / 64, Tt / 256, Bb), 256, 0, stream>>>(
      Qb, Kb, nullptr, S, Hh, Tt, 0, (long)Tt * Hh, (long)Tt * Hh,
      (long)Tt * Tt, 0.03125f);

  // 7) row softmax -> P (bf16)
  attn_softmax<<<Bb * Tt, 256, 0, stream>>>(S, P);

  // 8) ctx = P @ V  -> combined[:, 1024:2048] (bf16)
  gemm_bf16_wmma<1, 0, 0><<<dim3(Tt / 64, Hh / 256, Bb), 256, 0, stream>>>(
      P, Vt, nullptr, combined, Tt, 2 * Hh, Hh, (long)Tt * Tt, (long)Hh * Tt,
      (long)Tt * 2 * Hh, 1.f);

  // 9) combined[:, 0:1024] = gru_out (bf16)
  concat_gru<<<2048, 256, 0, stream>>>(gru_bf, combined);

  // 10) logits = combined @ W_fc^T + b_fc  -> d_out (fp32)
  gemm_bf16_wmma<0, 1, 0><<<dim3(MROWS / 64, Vv / 256, 1), 256, 0, stream>>>(
      combined, Wfc_bf, b_fc, out, 2 * Hh, Vv, 0, 0, 0, 0, 1.f);

  // 11) in-place log_softmax over V
  logsoftmax_inplace<<<MROWS, 256, 0, stream>>>(out);
}
